// CMA_encoder_34909494181959
// MI455X (gfx1250) — matrix operand order repaired and checked by reading, not compiled, not run on hardware
//
#include <hip/hip_runtime.h>
#include <math.h>
#include <stdint.h>

// ---------------- problem constants ----------------
#define BATCH   16
#define CDIM    384
#define NTOK    4096          // 64*64
#define NHEADS  4
#define HDIM    96
#define HIDDEN  1536
#define WIDTH_C 96            // CDIM/4

typedef __attribute__((ext_vector_type(16))) _Float16 v16h;
typedef __attribute__((ext_vector_type(8)))  _Float16 v8h;
typedef __attribute__((ext_vector_type(8)))  float    v8f;
typedef int v4i_vs __attribute__((vector_size(16)));   // matches builtin param type

// -------- CDNA5 async global->LDS (guarded; falls back to sync copy) --------
#if defined(__has_builtin)
#  if __has_builtin(__builtin_amdgcn_global_load_async_to_lds_b128)
#    define ASYNC_LDS 1
#  endif
#endif
#ifndef ASYNC_LDS
#  define ASYNC_LDS 0
#endif

#if ASYNC_LDS
#  if __has_builtin(__builtin_amdgcn_s_wait_asynccnt)
#    define WAIT_ASYNC() __builtin_amdgcn_s_wait_asynccnt(0)
#  else
#    define WAIT_ASYNC() asm volatile("s_wait_asynccnt 0x0" ::: "memory")
#  endif
__device__ __forceinline__ void async_copy_b128(const _Float16* src, _Float16* dstLds) {
  __builtin_amdgcn_global_load_async_to_lds_b128(
      (v4i_vs*)(uintptr_t)src,
      (__attribute__((address_space(3))) v4i_vs*)(uintptr_t)dstLds,
      0, 0);
}
#endif

// ---------------- WMMA GEMM: C = epilogue(A * B^T(+bias)) ----------------
// A: [M,K] f32 (or f16 when aHalf; then M%128==0, K%64==0 required).
// transA (f32 only): A[m][k]=Amem[k*lda+m].  B weight-style [N][K] f32
// (transB: Bt[n][k]=Bmem[k*ldb+n]).  Batched via blockIdx.z -> (zg,zh).
#define BM 128
#define BN 128
#define BKK 64
#define LDSA (BKK + 8)
#define LDSB (BKK + 8)

__device__ __forceinline__ v16h frag_a(const _Float16* tile, int kt, int lane) {
  int row = lane & 15;
  int k0  = kt * 32 + ((lane >> 4) << 3);          // 0 or 8
  const _Float16* p = tile + row * LDSA + k0;
  v8h lo = *(const v8h*)p;                          // K = k0..k0+7
  v8h hi = *(const v8h*)(p + 16);                   // K = k0+16..k0+23
  v16h r;
#pragma unroll
  for (int i = 0; i < 8; ++i) { r[i] = lo[i]; r[i + 8] = hi[i]; }
  return r;
}
__device__ __forceinline__ v16h frag_b(const _Float16* tile, int kt, int lane) {
  int col = lane & 15;
  int k0  = kt * 32 + ((lane >> 4) << 3);          // 0 or 8
  const _Float16* p = tile + col * LDSB + k0;
  v8h lo = *(const v8h*)p;
  v8h hi = *(const v8h*)(p + 16);
  v16h r;
#pragma unroll
  for (int i = 0; i < 8; ++i) { r[i] = lo[i]; r[i + 8] = hi[i]; }
  return r;
}

// mode: 0 = store (acc+bias); 1 = GELU(acc+bias);
//       2 = res[m][n] + scale[n]*(acc+bias)
//       3 = transposed: C[n*ldT+m] = res[n*ldT+m] + scale[n]*(acc+bias)
// outHalf: store f16 into (_Float16*)Cp (modes 0/1 only)
__global__ void __launch_bounds__(256)
gemm_wmma_f16(const void* __restrict__ Ap, long long sAg, long long sAh,
              const float* __restrict__ Bp, long long sBg, long long sBh,
              float* __restrict__ Cp, long long sCg, long long sCh,
              const float* __restrict__ bias,
              const float* __restrict__ resp, long long sRg, long long sRh,
              const float* __restrict__ scalep,
              int M, int Nn, int K, int lda, int ldb, int ldc,
              int transA, int transB, int mode, int nH, long long ldT,
              int aHalf, int outHalf)
{
  __shared__ __align__(16) _Float16 sA[BM * LDSA];
  __shared__ __align__(16) _Float16 sB[BN * LDSB];
  const int tid  = threadIdx.x;
  const int lane = tid & 31, wave = tid >> 5;
  const int wm = wave >> 1, wn = wave & 1;          // 4 x 2 waves
  const int zg = blockIdx.z / nH, zh = blockIdx.z % nH;
  const float*    A32 = aHalf ? nullptr : ((const float*)Ap + (long long)zg * sAg + (long long)zh * sAh);
  const _Float16* A16 = aHalf ? ((const _Float16*)Ap + (long long)zg * sAg + (long long)zh * sAh) : nullptr;
  const float* B = Bp + (long long)zg * sBg + (long long)zh * sBh;
  const long long coff = (long long)zg * sCg + (long long)zh * sCh;
  const long long roff = (long long)zg * sRg + (long long)zh * sRh;
  const int m0 = blockIdx.y * BM;
  const int n0 = blockIdx.x * BN;

  v8f acc[2][4];
#pragma unroll
  for (int i = 0; i < 2; ++i)
#pragma unroll
    for (int j = 0; j < 4; ++j)
#pragma unroll
      for (int e = 0; e < 8; ++e) acc[i][j][e] = 0.0f;

  for (int k0 = 0; k0 < K; k0 += BKK) {
    // ---- stage A tile ----
    if (aHalf) {
      // raw f16 tile, dims guaranteed in-range; 1024 x 16B chunks
#pragma unroll
      for (int it = 0; it < (BM * BKK) / (256 * 8); ++it) {      // 4 iters
        int idx = tid + it * 256;
        int r = idx >> 3;                 // 8 chunks per 64-wide row
        int c8 = (idx & 7) << 3;
        const _Float16* src = A16 + (long long)(m0 + r) * lda + (k0 + c8);
        _Float16* dst = sA + r * LDSA + c8;
#if ASYNC_LDS
        async_copy_b128(src, dst);
#else
        *(v8h*)dst = *(const v8h*)src;
#endif
      }
    } else if (!transA) {
      if (k0 + BKK < K)
        __builtin_prefetch(A32 + (long long)(m0 + (tid >> 1)) * lda + k0 + BKK, 0, 0);
#pragma unroll
      for (int it = 0; it < (BM * BKK) / 256; ++it) {
        int idx = tid + it * 256;
        int r = idx >> 6, c = idx & 63;
        int gm = m0 + r, gk = k0 + c;
        float v = (gm < M && gk < K) ? A32[(long long)gm * lda + gk] : 0.0f;
        sA[r * LDSA + c] = (_Float16)v;
      }
    } else {
#pragma unroll
      for (int it = 0; it < (BM * BKK) / 256; ++it) {
        int idx = tid + it * 256;
        int kk = idx >> 7, mm = idx & 127;
        int gm = m0 + mm, gk = k0 + kk;
        float v = (gm < M && gk < K) ? A32[(long long)gk * lda + gm] : 0.0f;
        sA[mm * LDSA + kk] = (_Float16)v;
      }
    }
    // ---- stage B tile (f32 weights -> f16) ----
    if (!transB) {
#pragma unroll
      for (int it = 0; it < (BN * BKK) / 256; ++it) {            // 32 iters
        int idx = tid + it * 256;
        int r = idx >> 6, c = idx & 63;
        int gn = n0 + r, gk = k0 + c;
        float v = (gn < Nn && gk < K) ? B[(long long)gn * ldb + gk] : 0.0f;
        sB[r * LDSB + c] = (_Float16)v;
      }
    } else {
#pragma unroll
      for (int it = 0; it < (BN * BKK) / 256; ++it) {
        int idx = tid + it * 256;
        int kk = idx >> 7, nn = idx & 127;
        int gn = n0 + nn, gk = k0 + kk;
        float v = (gn < Nn && gk < K) ? B[(long long)gk * ldb + gn] : 0.0f;
        sB[nn * LDSB + kk] = (_Float16)v;
      }
    }
#if ASYNC_LDS
    if (aHalf) WAIT_ASYNC();
#endif
    __syncthreads();
#pragma unroll
    for (int kt = 0; kt < 2; ++kt) {
      v16h a0 = frag_a(sA + (wm * 32) * LDSA, kt, lane);
      v16h a1 = frag_a(sA + (wm * 32 + 16) * LDSA, kt, lane);
      v16h b0 = frag_b(sB + (wn * 64) * LDSB, kt, lane);
      v16h b1 = frag_b(sB + (wn * 64 + 16) * LDSB, kt, lane);
      v16h b2 = frag_b(sB + (wn * 64 + 32) * LDSB, kt, lane);
      v16h b3 = frag_b(sB + (wn * 64 + 48) * LDSB, kt, lane);
      acc[0][0] = __builtin_amdgcn_wmma_f32_16x16x32_f16(false, a0, false, b0, (short)0, acc[0][0], false, false);
      acc[0][1] = __builtin_amdgcn_wmma_f32_16x16x32_f16(false, a0, false, b1, (short)0, acc[0][1], false, false);
      acc[0][2] = __builtin_amdgcn_wmma_f32_16x16x32_f16(false, a0, false, b2, (short)0, acc[0][2], false, false);
      acc[0][3] = __builtin_amdgcn_wmma_f32_16x16x32_f16(false, a0, false, b3, (short)0, acc[0][3], false, false);
      acc[1][0] = __builtin_amdgcn_wmma_f32_16x16x32_f16(false, a1, false, b0, (short)0, acc[1][0], false, false);
      acc[1][1] = __builtin_amdgcn_wmma_f32_16x16x32_f16(false, a1, false, b1, (short)0, acc[1][1], false, false);
      acc[1][2] = __builtin_amdgcn_wmma_f32_16x16x32_f16(false, a1, false, b2, (short)0, acc[1][2], false, false);
      acc[1][3] = __builtin_amdgcn_wmma_f32_16x16x32_f16(false, a1, false, b3, (short)0, acc[1][3], false, false);
    }
    __syncthreads();
  }

  // ---- epilogue ----
  const int lhalf = (lane >> 4) << 3;   // 0 | 8
  const int lcol  = lane & 15;
#pragma unroll
  for (int i = 0; i < 2; ++i) {
#pragma unroll
    for (int j = 0; j < 4; ++j) {
      int tm = m0 + wm * 32 + i * 16;
      int tn = n0 + wn * 64 + j * 16;
      int n = tn + lcol;
      if (n >= Nn) continue;
      float bv = bias   ? bias[n]   : 0.0f;
      float sv = scalep ? scalep[n] : 1.0f;
#pragma unroll
      for (int v = 0; v < 8; ++v) {
        int m = tm + lhalf + v;
        if (m >= M) continue;
        float xv = acc[i][j][v] + bv;
        if (mode == 1) xv = 0.5f * xv * (1.0f + erff(xv * 0.7071067811865475f));
        if (outHalf) {
          ((_Float16*)Cp)[coff + (long long)m * ldc + n] = (_Float16)xv;
        } else if (mode == 2) {
          Cp[coff + (long long)m * ldc + n] = resp[roff + (long long)m * ldc + n] + sv * xv;
        } else if (mode == 3) {
          long long a = (long long)n * ldT + m;
          Cp[coff + a] = resp[roff + a] + sv * xv;
        } else {
          Cp[coff + (long long)m * ldc + n] = xv;
        }
      }
    }
  }
}

// ---------------- depthwise 3x3 (zero pad), optional input add ----------------
__global__ void dwconv3x3_kernel(const float* __restrict__ inPrev,
                                 const float* __restrict__ inAdd,
                                 const float* __restrict__ w9,
                                 const float* __restrict__ bias,
                                 float* __restrict__ out)
{
  long long idx = (long long)blockIdx.x * 256 + threadIdx.x;   // B*96*4096
  if (idx >= (long long)BATCH * WIDTH_C * NTOK) return;
  int n = (int)(idx & 4095);
  int c = (int)((idx >> 12) % WIDTH_C);
  int b = (int)(idx / ((long long)WIDTH_C * NTOK));
  int h = n >> 6, wx = n & 63;
  long long bOff = (long long)b * CDIM * NTOK;
  const float* ip = inPrev + bOff + (long long)c * NTOK;
  const float* ia = inAdd ? inAdd + bOff + (long long)c * NTOK : nullptr;
  float acc = bias[c];
#pragma unroll
  for (int dy = 0; dy < 3; ++dy)
#pragma unroll
    for (int dx = 0; dx < 3; ++dx) {
      int yy = h + dy - 1, xx = wx + dx - 1;
      if (yy >= 0 && yy < 64 && xx >= 0 && xx < 64) {
        float v = ip[yy * 64 + xx];
        if (ia) v += ia[yy * 64 + xx];
        acc += v * w9[c * 9 + dy * 3 + dx];
      }
    }
  out[bOff + (long long)c * NTOK + n] = acc;
}

// ---------------- Fourier position feature matrix [4096,64] ----------------
__global__ void posmat_kernel(float* __restrict__ pm)
{
  int idx = blockIdx.x * 256 + threadIdx.x;          // 4096*64
  if (idx >= NTOK * 64) return;
  int j = idx & 63, n = idx >> 6;
  int h = n >> 6, w = n & 63;
  const float scale = 6.283185307179586f;
  float coord = (j < 32) ? ((float)(h + 1) * scale / 64.000001f)
                         : ((float)(w + 1) * scale / 64.000001f);
  int jj = j & 31;
  float dimt = powf(10000.0f, (float)(2 * (jj >> 1)) * (1.0f / 32.0f));
  float a = coord / dimt;
  pm[idx] = (jj & 1) ? cosf(a) : sinf(a);
}

// ---------------- x3[b,n,c] = conv(x)|x  transposed + pos[n,c] ----------------
__global__ void build_x3_kernel(const float* __restrict__ xconv,
                                const float* __restrict__ x,
                                const float* __restrict__ pos,
                                float* __restrict__ x3)
{
  long long idx = (long long)blockIdx.x * 256 + threadIdx.x;   // B*N*C
  if (idx >= (long long)BATCH * NTOK * CDIM) return;
  int c = (int)(idx % CDIM);
  long long bn = idx / CDIM;
  int n = (int)(bn % NTOK);
  int b = (int)(bn / NTOK);
  const float* s = (c < 3 * WIDTH_C) ? xconv : x;
  x3[idx] = s[((long long)b * CDIM + c) * NTOK + n] + pos[n * CDIM + c];
}

// ---------------- LayerNorm rows of 384 -> f16 (one wave per row) ----------------
__global__ void layernorm_kernel(const float* __restrict__ in, _Float16* __restrict__ out,
                                 const float* __restrict__ w, const float* __restrict__ b,
                                 int rows, int foldMode)
{
  int wave = threadIdx.x >> 5, lane = threadIdx.x & 31;
  int row = blockIdx.x * 8 + wave;
  if (row >= rows) return;
  long long src = row;
  if (foldMode) {
    int bb = row >> 12;
    int pix = row & 4095;
    int hh = pix >> 6, ww = pix & 63;
    int g = bb * 4 + (hh & 1) * 2 + (ww & 1);
    int t = (hh >> 1) * 32 + (ww >> 1);
    src = (long long)g * 1024 + t;
  }
  const float* ip = in + src * CDIM;
  float vals[12];
  float s = 0.0f;
#pragma unroll
  for (int i = 0; i < 12; ++i) { vals[i] = ip[lane + i * 32]; s += vals[i]; }
#pragma unroll
  for (int m = 16; m > 0; m >>= 1) s += __shfl_xor(s, m, 32);
  float mean = s * (1.0f / CDIM);
  float vs = 0.0f;
#pragma unroll
  for (int i = 0; i < 12; ++i) { float d = vals[i] - mean; vs += d * d; }
#pragma unroll
  for (int m = 16; m > 0; m >>= 1) vs += __shfl_xor(vs, m, 32);
  float inv = rsqrtf(vs * (1.0f / CDIM) + 1e-6f);
  _Float16* op = out + (long long)row * CDIM;
#pragma unroll
  for (int i = 0; i < 12; ++i) {
    int c = lane + i * 32;
    op[c] = (_Float16)((vals[i] - mean) * inv * w[c] + b[c]);
  }
}

// ---------------- split KV + L2-normalize K rows over tokens ----------------
__global__ void split_kv_l2norm(const float* __restrict__ kv,
                                float* __restrict__ kO, float* __restrict__ vO,
                                int Ntok)
{
  int r = blockIdx.x;                               // G*H*hd rows
  int g = r / (NHEADS * HDIM);
  int rem = r % (NHEADS * HDIM);
  int h = rem / HDIM, d = rem % HDIM;
  const float* base = kv + (long long)g * Ntok * (2 * CDIM);
  int ck = h * HDIM + d;
  int cv = CDIM + h * HDIM + d;
  __shared__ float rowk[4096];
  __shared__ float red[256];
  float ss = 0.0f;
  for (int n = threadIdx.x; n < Ntok; n += 256) {
    float kvv = base[(long long)n * (2 * CDIM) + ck];
    rowk[n] = kvv; ss += kvv * kvv;
    vO[(long long)r * Ntok + n] = base[(long long)n * (2 * CDIM) + cv];
  }
  red[threadIdx.x] = ss; __syncthreads();
  for (int s = 128; s > 0; s >>= 1) {
    if ((int)threadIdx.x < s) red[threadIdx.x] += red[threadIdx.x + s];
    __syncthreads();
  }
  float inv = 1.0f / fmaxf(sqrtf(red[0]), 1e-12f);
  for (int n = threadIdx.x; n < Ntok; n += 256)
    kO[(long long)r * Ntok + n] = rowk[n] * inv;
}

// ---------------- XCA softmax: softmax(logits*temp[h]) over 96 ----------------
__global__ void softmax_xca_kernel(float* __restrict__ lg, const float* __restrict__ temp,
                                   int rowsTotal)
{
  int wave = threadIdx.x >> 5, lane = threadIdx.x & 31;
  int row = blockIdx.x * 8 + wave;
  if (row >= rowsTotal) return;
  float tp = temp[(row / HDIM) & (NHEADS - 1)];
  float* p = lg + (long long)row * HDIM;
  float v0 = p[lane] * tp, v1 = p[lane + 32] * tp, v2 = p[lane + 64] * tp;
  float mx = fmaxf(v0, fmaxf(v1, v2));
#pragma unroll
  for (int m = 16; m > 0; m >>= 1) mx = fmaxf(mx, __shfl_xor(mx, m, 32));
  float e0 = expf(v0 - mx), e1 = expf(v1 - mx), e2 = expf(v2 - mx);
  float s = e0 + e1 + e2;
#pragma unroll
  for (int m = 16; m > 0; m >>= 1) s += __shfl_xor(s, m, 32);
  float inv = 1.0f / s;
  p[lane] = e0 * inv; p[lane + 32] = e1 * inv; p[lane + 64] = e2 * inv;
}

// ------ window softmax: softmax(0.5*logits/sqrt(96) + 0.5*softmax(logits)) ------
__global__ void softmax_wa_kernel(float* __restrict__ lg, int rowsTotal)
{
  int wave = threadIdx.x >> 5, lane = threadIdx.x & 31;
  int row = blockIdx.x * 8 + wave;
  if (row >= rowsTotal) return;
  float* p = lg + (long long)row * HDIM;
  float v0 = p[lane], v1 = p[lane + 32], v2 = p[lane + 64];
  float mx = fmaxf(v0, fmaxf(v1, v2));
#pragma unroll
  for (int m = 16; m > 0; m >>= 1) mx = fmaxf(mx, __shfl_xor(mx, m, 32));
  float e0 = expf(v0 - mx), e1 = expf(v1 - mx), e2 = expf(v2 - mx);
  float s = e0 + e1 + e2;
#pragma unroll
  for (int m = 16; m > 0; m >>= 1) s += __shfl_xor(s, m, 32);
  float inv = 1.0f / s;
  const float c = 0.10206207261596577f;             // 1/sqrt(96)
  float a0 = 0.5f * v0 * c + 0.5f * e0 * inv;
  float a1 = 0.5f * v1 * c + 0.5f * e1 * inv;
  float a2 = 0.5f * v2 * c + 0.5f * e2 * inv;
  float mx2 = fmaxf(a0, fmaxf(a1, a2));
#pragma unroll
  for (int m = 16; m > 0; m >>= 1) mx2 = fmaxf(mx2, __shfl_xor(mx2, m, 32));
  float f0 = expf(a0 - mx2), f1 = expf(a1 - mx2), f2 = expf(a2 - mx2);
  float s2 = f0 + f1 + f2;
#pragma unroll
  for (int m = 16; m > 0; m >>= 1) s2 += __shfl_xor(s2, m, 32);
  float inv2 = 1.0f / s2;
  p[lane] = f0 * inv2; p[lane + 32] = f1 * inv2; p[lane + 64] = f2 * inv2;
}

// ---------------- unfold to [64,1024,384]; mode0 from [B,N,C] with scale,
//                  mode1 from [B,C,N]; optional f16 output ----------------
__global__ void unfold_kernel(const float* __restrict__ src, float* __restrict__ dst,
                              const float* __restrict__ sw, const float* __restrict__ sb,
                              int mode, int outHalf)
{
  long long idx = (long long)blockIdx.x * 256 + threadIdx.x;   // 64*1024*384
  if (idx >= 64LL * 1024 * CDIM) return;
  int c = (int)(idx % CDIM);
  long long gt = idx / CDIM;
  int t = (int)(gt & 1023);
  int g = (int)(gt >> 10);
  int b = g >> 2, pp = g & 3, ph = pp >> 1, pw = pp & 1;
  int yh = t >> 5, xw = t & 31;
  int npix = (yh * 2 + ph) * 64 + (xw * 2 + pw);
  float v;
  if (mode == 0) v = src[((long long)b * NTOK + npix) * CDIM + c] * sw[c] + sb[c];
  else           v = src[((long long)b * CDIM + c) * NTOK + npix];
  if (outHalf) ((_Float16*)dst)[idx] = (_Float16)v;
  else         dst[idx] = v;
}

// ---------------- host side ----------------
static void launch_gemm(hipStream_t s,
                        const void* A, long long sAg, long long sAh,
                        const float* B, long long sBg, long long sBh,
                        float* C, long long sCg, long long sCh,
                        const float* bias,
                        const float* res, long long sRg, long long sRh,
                        const float* scale,
                        int M, int N, int K, int lda, int ldb, int ldc,
                        int tA, int tB, int mode, int nH, long long ldT, int Z,
                        int aHalf, int outHalf)
{
  dim3 grid((N + BN - 1) / BN, (M + BM - 1) / BM, Z);
  gemm_wmma_f16<<<grid, 256, 0, s>>>(A, sAg, sAh, B, sBg, sBh, C, sCg, sCh,
                                     bias, res, sRg, sRh, scale,
                                     M, N, K, lda, ldb, ldc, tA, tB, mode, nH, ldT,
                                     aHalf, outHalf);
}

extern "C" void kernel_launch(void* const* d_in, const int* in_sizes, int n_in,
                              void* d_out, int out_size, void* d_ws, size_t ws_size,
                              hipStream_t stream)
{
  const float* x          = (const float*)d_in[0];
  const float* convs_w    = (const float*)d_in[1];
  const float* convs_b    = (const float*)d_in[2];
  const float* pos_w      = (const float*)d_in[3];
  const float* pos_b      = (const float*)d_in[4];
  const float* ln_xca_w   = (const float*)d_in[5];
  const float* ln_xca_b   = (const float*)d_in[6];
  const float* gamma_xca  = (const float*)d_in[7];
  const float* xca_temp   = (const float*)d_in[8];
  const float* xca_kv_w   = (const float*)d_in[9];
  const float* xca_kv_b   = (const float*)d_in[10];
  const float* xca_proj_w = (const float*)d_in[11];
  const float* xca_proj_b = (const float*)d_in[12];
  const float* conv_out_w = (const float*)d_in[13];
  const float* conv_out_b = (const float*)d_in[14];
  const float* wa_kv_w    = (const float*)d_in[15];
  const float* wa_kv_b    = (const float*)d_in[16];
  const float* wa_proj_w  = (const float*)d_in[17];
  const float* wa_proj_b  = (const float*)d_in[18];
  const float* ln_w       = (const float*)d_in[19];
  const float* ln_b       = (const float*)d_in[20];
  const float* pw1_w      = (const float*)d_in[21];
  const float* pw1_b      = (const float*)d_in[22];
  const float* pw2_w      = (const float*)d_in[23];
  const float* pw2_b      = (const float*)d_in[24];
  const float* gamma      = (const float*)d_in[25];
  float* out = (float*)d_out;

  float* ws = (float*)d_ws;
  size_t off = 0;
  auto take = [&](size_t n) { float* p = ws + off; off += n; return p; };
  const size_t SZ = (size_t)BATCH * CDIM * NTOK;                 // 25,165,824 floats
  float* buf_xconv = take(SZ);            // -> reused as x2p (f32)
  float* buf_x3    = take(SZ);
  float* buf_ln    = take(SZ);            // constu (f32)
  float* buf_kv    = take(2 * SZ);        // [rows, 768] f32
  float* buf_kt    = take(SZ);
  float* buf_vt    = take(SZ);
  float* buf_ctx   = take(SZ);            // XCA context f32
  float* buf_x4    = take(SZ);
  float* buf_logit = take((size_t)256 * HDIM * HDIM);
  float* buf_posm  = take((size_t)NTOK * 64);
  float* buf_pos   = take((size_t)NTOK * CDIM);
  _Float16* h_ln   = (_Float16*)take(SZ / 2);     // LN output f16 (reused for final LN)
  _Float16* h_xsu  = (_Float16*)take(SZ / 2);     // unfolded scaled x, f16
  _Float16* h_ctx2 = (_Float16*)take(SZ / 2);     // window ctx (torch-permuted), f16
  _Float16* h_hid  = (_Float16*)take(2 * SZ);     // MLP hidden f16 [65536,1536]
  (void)ws_size; (void)in_sizes; (void)n_in; (void)out_size;

  const long long GHN  = (long long)HDIM * NTOK;                 // 393216
  const long long BNC  = (long long)NTOK * CDIM;                 // 1572864
  const int dwGrid = (BATCH * WIDTH_C * NTOK + 255) / 256;

  // 1) Res2Net depthwise conv chain
  dwconv3x3_kernel<<<dwGrid, 256, 0, stream>>>(x, nullptr, convs_w, convs_b, buf_xconv);
  dwconv3x3_kernel<<<dwGrid, 256, 0, stream>>>(buf_xconv, x + (size_t)WIDTH_C * NTOK,
      convs_w + 1 * WIDTH_C * 9, convs_b + 1 * WIDTH_C, buf_xconv + (size_t)WIDTH_C * NTOK);
  dwconv3x3_kernel<<<dwGrid, 256, 0, stream>>>(buf_xconv + (size_t)WIDTH_C * NTOK,
      x + (size_t)2 * WIDTH_C * NTOK,
      convs_w + 2 * WIDTH_C * 9, convs_b + 2 * WIDTH_C, buf_xconv + (size_t)2 * WIDTH_C * NTOK);

  // 2) pos features + pos projection (WMMA GEMM)
  posmat_kernel<<<(NTOK * 64) / 256, 256, 0, stream>>>(buf_posm);
  launch_gemm(stream, buf_posm, 0, 0, pos_w, 0, 0, buf_pos, 0, 0, pos_b,
              nullptr, 0, 0, nullptr, NTOK, CDIM, 64, 64, 64, CDIM, 0, 0, 0, 1, 0, 1, 0, 0);

  // 3) x3 = transpose(conv out) + pos
  build_x3_kernel<<<(int)(SZ / 256), 256, 0, stream>>>(buf_xconv, x, buf_pos, buf_x3);

  // 4) LN for XCA -> f16
  layernorm_kernel<<<BATCH * NTOK / 8, 256, 0, stream>>>(buf_x3, h_ln, ln_xca_w, ln_xca_b,
                                                         BATCH * NTOK, 0);
  // 5) XCA KV projection (f16 A, async staged)
  launch_gemm(stream, h_ln, 0, 0, xca_kv_w, 0, 0, buf_kv, 0, 0, xca_kv_b,
              nullptr, 0, 0, nullptr, BATCH * NTOK, 2 * CDIM, CDIM, CDIM, CDIM, 2 * CDIM,
              0, 0, 0, 1, 0, 1, 1, 0);
  // 6) split K/V + L2norm(K) over N
  split_kv_l2norm<<<BATCH * NHEADS * HDIM, 256, 0, stream>>>(buf_kv, buf_kt, buf_vt, NTOK);
  // 7) XCA logits: q (= input slice) . k^T, batched over b*h
  launch_gemm(stream, x, GHN, 0, buf_kt, GHN, 0, buf_logit, HDIM * HDIM, 0, nullptr,
              nullptr, 0, 0, nullptr, HDIM, HDIM, NTOK, NTOK, NTOK, HDIM,
              0, 0, 0, 1, 0, BATCH * NHEADS, 0, 0);
  softmax_xca_kernel<<<BATCH * NHEADS * HDIM / 8, 256, 0, stream>>>(buf_logit, xca_temp,
                                                                    BATCH * NHEADS * HDIM);
  // 8) XCA context: attn @ v  -> [b, c, n]
  launch_gemm(stream, buf_logit, HDIM * HDIM, 0, buf_vt, GHN, 0, buf_ctx, GHN, 0, nullptr,
              nullptr, 0, 0, nullptr, HDIM, NTOK, HDIM, HDIM, NTOK, NTOK,
              0, 1, 0, 1, 0, BATCH * NHEADS, 0, 0);
  // 9) XCA proj + layer-scale residual: x4 = x3 + gamma_xca*(ctx^T @ W^T + b)
  launch_gemm(stream, buf_ctx, BNC, 0, xca_proj_w, 0, 0, buf_x4, BNC, 0, xca_proj_b,
              buf_x3, BNC, 0, gamma_xca, NTOK, CDIM, CDIM, NTOK, CDIM, CDIM,
              1, 0, 2, 1, 0, BATCH, 0, 0);

  // 10) unfold: xsu = unfold(conv_out(x4)) -> f16,  constu = unfold(input) -> f32
  unfold_kernel<<<(int)(SZ / 256), 256, 0, stream>>>(buf_x4, (float*)h_xsu, conv_out_w, conv_out_b, 0, 1);
  unfold_kernel<<<(int)(SZ / 256), 256, 0, stream>>>(x, buf_ln, nullptr, nullptr, 1, 0);

  // 11) window KV projection (f16 A)
  launch_gemm(stream, h_xsu, 0, 0, wa_kv_w, 0, 0, buf_kv, 0, 0, wa_kv_b,
              nullptr, 0, 0, nullptr, 64 * 1024, 2 * CDIM, CDIM, CDIM, CDIM, 2 * CDIM,
              0, 0, 0, 1, 0, 1, 1, 0);
  split_kv_l2norm<<<64 * NHEADS * HDIM, 256, 0, stream>>>(buf_kv, buf_kt, buf_vt, 1024);
  // 12) window logits: q=constu^T (per g,h) . k^T
  launch_gemm(stream, buf_ln, (long long)1024 * CDIM, HDIM,
              buf_kt, (long long)NHEADS * HDIM * 1024, (long long)HDIM * 1024,
              buf_logit, (long long)NHEADS * HDIM * HDIM, HDIM * HDIM, nullptr,
              nullptr, 0, 0, nullptr, HDIM, HDIM, 1024, CDIM, 1024, HDIM,
              1, 0, 0, NHEADS, 0, 64 * NHEADS, 0, 0);
  softmax_wa_kernel<<<64 * NHEADS * HDIM / 8, 256, 0, stream>>>(buf_logit, 64 * NHEADS * HDIM);
  // 13) window context -> permuted [h][g][d][t] f16 (reproduces torch reshape)
  launch_gemm(stream, buf_logit, (long long)NHEADS * HDIM * HDIM, HDIM * HDIM,
              buf_vt, (long long)NHEADS * HDIM * 1024, (long long)HDIM * 1024,
              (float*)h_ctx2, (long long)HDIM * 1024, (long long)64 * HDIM * 1024, nullptr,
              nullptr, 0, 0, nullptr, HDIM, 1024, HDIM, HDIM, 1024, 1024,
              0, 1, 0, NHEADS, 0, 64 * NHEADS, 0, 1);
  // 14) window proj (f16 A viewed as [65536,384])
  launch_gemm(stream, h_ctx2, 0, 0, wa_proj_w, 0, 0, buf_xconv, 0, 0, wa_proj_b,
              nullptr, 0, 0, nullptr, 64 * 1024, CDIM, CDIM, CDIM, CDIM, CDIM,
              0, 0, 0, 1, 0, 1, 1, 0);
  // 15) fold + LayerNorm -> f16
  layernorm_kernel<<<BATCH * NTOK / 8, 256, 0, stream>>>(buf_xconv, h_ln, ln_w, ln_b,
                                                         BATCH * NTOK, 1);
  // 16) MLP pw1 + exact GELU -> f16 hidden
  launch_gemm(stream, h_ln, 0, 0, pw1_w, 0, 0, (float*)h_hid, 0, 0, pw1_b,
              nullptr, 0, 0, nullptr, BATCH * NTOK, HIDDEN, CDIM, CDIM, CDIM, HIDDEN,
              0, 0, 1, 1, 0, 1, 1, 1);
  // 17) MLP pw2 + gamma layer-scale + transposed residual write to NCHW output
  launch_gemm(stream, h_hid, (long long)NTOK * HIDDEN, 0, pw2_w, 0, 0,
              out, BNC, 0, pw2_b, x, BNC, 0, gamma,
              NTOK, CDIM, HIDDEN, HIDDEN, HIDDEN, 0,
              0, 0, 3, 1, NTOK, BATCH, 1, 0);
}
